// SequenceClassifierRNN_69578470195540
// MI455X (gfx1250) — matrix operand order repaired and checked by reading, not compiled, not run on hardware
//
#include <hip/hip_runtime.h>
#include <hip/hip_bf16.h>

// CDNA5 fused RNN: one workgroup per 16-batch tile, hidden state carried in
// LDS across all T steps. bf16 WMMA (16x16x32) with f32 accumulate.
// - Both weight matrices live in VGPRs (loop-invariant B-fragments).
// - Input projection xp_{t+1} computed as an independent WMMA chain while the
//   serial recurrent chain for step t runs (critical path = 8 chained WMMAs).
// - Bias folded into the xp accumulator init (no adds on the tanh path).
// - x tiles streamed with GLOBAL_LOAD_ASYNC_TO_LDS_B128 into a 7-deep f32 LDS
//   ring (ASYNCcnt-tracked, ~3.5 steps of HBM latency hidden); each tile is
//   converted to bf16 once, cooperatively (4 cvt_pk per thread), into a bf16
//   ping-pong consumed as WMMA fragments.
// - One barrier per step.

typedef __attribute__((ext_vector_type(16))) __bf16 v16bf;
typedef __attribute__((ext_vector_type(8)))  __bf16 v8bf;
typedef __attribute__((ext_vector_type(16))) float  v16f;
typedef __attribute__((ext_vector_type(8)))  float  v8f;

#define T_STEPS 2048
#define IDIM    256
#define HDIM    256
#define ODIM    16
#define HSTRIDE 264   // bf16 elements per row (528 B = 33*16: bank-spread, 16B aligned)
#define XSTRIDE 256   // f32 elements per x-ring row (thread-contiguous access; no pad needed)
#define XRING   7     // f32 x tile ring depth

union FragU { uint4 q[2]; v16bf v; };

// ---- A-fragment (16x32 bf16, M x K) from a bf16 tile ----
// lane lm = row M; lanes 0-15 hold K {k0..k0+15},
// lanes 16-31 hold K {k0+16..k0+31}  (matches B-fragment order).
__device__ __forceinline__ v16bf lds_frag_a(const __bf16* tile, int lm, int hi, int k0) {
  const uint4* p = (const uint4*)(tile + lm * HSTRIDE + k0 + hi * 16);
  FragU f;
  f.q[0] = p[0];   // K = k0+hi*16 .. +7
  f.q[1] = p[1];   // K = k0+hi*16+8 .. +15
  return f.v;
}

// ---- A-fragment from an f32 tile (prologue only) ----
__device__ __forceinline__ v16bf lds_frag_a_f32(const float* tile, int lm, int hi, int k0) {
  const float4* p = (const float4*)(tile + lm * XSTRIDE + k0 + hi * 16);
  const float4 f0 = p[0];
  const float4 f1 = p[1];
  const float4 f2 = p[2];
  const float4 f3 = p[3];
  v16f t = {f0.x, f0.y, f0.z, f0.w, f1.x, f1.y, f1.z, f1.w,
            f2.x, f2.y, f2.z, f2.w, f3.x, f3.y, f3.z, f3.w};
  return __builtin_convertvector(t, v16bf);
}

__device__ __forceinline__ float fast_tanh(float x) {
#if __has_builtin(__builtin_amdgcn_tanhf)
  return __builtin_amdgcn_tanhf(x);      // v_tanh_f32 (gfx1250 trans op)
#elif __has_builtin(__builtin_amdgcn_tanh_f32)
  return __builtin_amdgcn_tanh_f32(x);
#else
  return tanhf(x);
#endif
}

// 32-bit raw LDS byte address (generic -> addrspace(3) -> int).
__device__ __forceinline__ unsigned lds_addr_u32(const void* p) {
  return (unsigned)(unsigned long long)(__attribute__((address_space(3))) const void*)p;
}

// One 16-byte async global->LDS copy per active lane (ASYNCcnt-tracked).
__device__ __forceinline__ void async_copy_b128(const float* gsrc, const float* ldst) {
  asm volatile("global_load_async_to_lds_b128 %0, %1, off"
               :: "v"(lds_addr_u32(ldst)), "v"(gsrc)
               : "memory");
}

// Convert 8 f32 (2x ds_load_b128) -> 8 bf16 (4x v_cvt_pk_bf16_f32) -> 1 ds_store_b128.
__device__ __forceinline__ void convert_tile_slice(const float* src, __bf16* dst) {
  const float4 a = *(const float4*)src;
  const float4 b = *(const float4*)(src + 4);
  v8f f = {a.x, a.y, a.z, a.w, b.x, b.y, b.z, b.w};
  *(v8bf*)dst = __builtin_convertvector(f, v8bf);
}

__global__ __launch_bounds__(512) void rnn_fused_kernel(
    const float* __restrict__ xg,     // [B, T, I]
    const float* __restrict__ wih_g,  // [I, H]
    const float* __restrict__ whh_g,  // [H, H]
    const float* __restrict__ bh,     // [H]
    const float* __restrict__ fcw,    // [O, H]
    const float* __restrict__ fcb,    // [O]
    float* __restrict__ out)          // [B, O]
{
  __shared__ __align__(16) __bf16 sH[2][16 * HSTRIDE];      // hidden ping-pong, 16.5 KB
  __shared__ __align__(16) __bf16 sXb[2][16 * HSTRIDE];     // bf16 x ping-pong, 16.5 KB
  __shared__ __align__(16) float  sXf[XRING][16 * XSTRIDE]; // f32 x ring, 112 KB

  const int tid  = threadIdx.x;
  const int wave = tid >> 5;
  const int lane = tid & 31;
  const int lm   = lane & 15;
  const int hi   = lane >> 4;
  const int n0   = wave * 16;          // this wave's output-column tile of H
  const int bb   = blockIdx.x * 16;    // batch-row base

  // ---- zero h_0 ----
  for (int idx = tid; idx < 16 * HSTRIDE; idx += 512) {
    sH[0][idx] = (__bf16)0.0f;
    sH[1][idx] = (__bf16)0.0f;
  }

  // ---- both weight matrices resident in registers (B-fragment layout) ----
  // B-frag element j, lane L: W[k][n] with k = ks*32 + hi*16 + j, n = n0 + lm.
  v16bf whh[8], wihf[8];
#pragma unroll
  for (int ks = 0; ks < 8; ++ks) {
#pragma unroll
    for (int j = 0; j < 16; ++j) {
      const int k = ks * 32 + hi * 16 + j;
      whh[ks][j]  = (__bf16)whh_g[k * HDIM + (n0 + lm)];
      wihf[ks][j] = (__bf16)wih_g[k * HDIM + (n0 + lm)];
    }
  }
  const float bias = bh[n0 + lm];
  const v8f biasv = {bias, bias, bias, bias, bias, bias, bias, bias};

  // ---- x staging geometry: 512 threads x 8 floats = one 16x256 tile ----
  const int xm  = tid >> 5;         // tile row 0..15
  const int xcf = (tid & 31) * 8;   // column base (floats)
  const float* xbase = xg + (size_t)(bb + xm) * T_STEPS * IDIM + xcf;

  // ---- prologue: fill f32 ring with tiles x_0 .. x_5 (slots 0..5) ----
#pragma unroll
  for (int j = 0; j < XRING - 1; ++j) {
    const float* g = xbase + (size_t)j * IDIM;
    float* l = &sXf[j][xm * XSTRIDE + xcf];
    async_copy_b128(g, l);
    async_copy_b128(g + 4, l + 4);
  }
  asm volatile("s_wait_asynccnt 0" ::: "memory");
  __syncthreads();   // h_0 and x_0..x_5 (f32) visible

  // convert x_1 -> sXb[0]; compute accX = b_h + x_0 @ W_ih[:, n0:n0+16]
  convert_tile_slice(&sXf[1][xm * XSTRIDE + xcf], &sXb[0][xm * HSTRIDE + xcf]);
  v8f accX = biasv;
#pragma unroll
  for (int ks = 0; ks < 8; ++ks) {
    v16bf a = lds_frag_a_f32(sXf[0], lm, hi, ks * 32);
    accX = __builtin_amdgcn_wmma_f32_16x16x32_bf16(false, a, false, wihf[ks],
                                                   (short)0, accX, false, false);
  }
  __syncthreads();   // sXb[0] = x_1 visible; slot-0 reads done

  // ---- sequential recurrence over T ----
  // Invariant at top of iteration t (p = t&1):
  //   accX = b_h + xp_t, sH[p] = h_t, sXb[p] = x_{t+1} (bf16),
  //   f32 ring holds (or is receiving) tiles x_t .. x_{t+5} (tile j in slot j mod 7).
  int sIssue = XRING - 1;   // slot of tile t+6 (t=0 -> 6)
  int sConv  = 2;           // slot of tile t+2 (t=0 -> 2)
#pragma unroll 2
  for (int t = 0; t < T_STEPS; ++t) {
    const int p = t & 1;

    // ---- issue async copy of tile x_{t+6} into slot (t+6) mod 7.
    // That slot held x_{t-1}, whose f32 reads finished at iter t-3. Safe.
    if (t + XRING - 1 < T_STEPS) {
      const float* g = xbase + (size_t)(t + XRING - 1) * IDIM;
      float* l = &sXf[sIssue][xm * XSTRIDE + xcf];
      async_copy_b128(g, l);
      async_copy_b128(g + 4, l + 4);
    }

    // ---- cooperative one-time conversion: f32 x_{t+2} -> bf16 sXb[1-p]
    if (t + 2 < T_STEPS) {
      convert_tile_slice(&sXf[sConv][xm * XSTRIDE + xcf],
                         &sXb[1 - p][xm * HSTRIDE + xcf]);
    }

    // ---- critical chain: accR = (b_h + xp_t) + h_t @ W_hh[:, n0:n0+16]
    v8f accR = accX;
    const __bf16* hb = sH[p];
#pragma unroll
    for (int ks = 0; ks < 8; ++ks) {
      v16bf a = lds_frag_a(hb, lm, hi, ks * 32);
      accR = __builtin_amdgcn_wmma_f32_16x16x32_bf16(false, a, false, whh[ks],
                                                     (short)0, accR, false, false);
    }

    // ---- off-critical-path: b_h + xp_{t+1} (independent acc chain, bias-seeded)
    v8f accN = biasv;
    if (t + 1 < T_STEPS) {
      const __bf16* xb = sXb[p];
#pragma unroll
      for (int ks = 0; ks < 8; ++ks) {
        v16bf a = lds_frag_a(xb, lm, hi, ks * 32);
        accN = __builtin_amdgcn_wmma_f32_16x16x32_bf16(false, a, false, wihf[ks],
                                                       (short)0, accN, false, false);
      }
    }

    // ---- tanh, write h_{t+1} (C/D layout: VGPR r -> row r+8*hi, col n0+lm)
    __bf16* hn = sH[1 - p];
#pragma unroll
    for (int r = 0; r < 8; ++r) {
      const float v = fast_tanh(accR[r]);
      hn[(r + 8 * hi) * HSTRIDE + (n0 + lm)] = (__bf16)v;
    }

    // ---- allow 3 younger issue-batches (2 copies each) in flight; in-order
    // completion => tile x_{t+3} (f32-read next iter) has landed.
    asm volatile("s_wait_asynccnt 6" ::: "memory");
    __syncthreads();   // publish h_{t+1}, bf16 x_{t+2}, landed f32 tiles
    accX = accN;

    // rotate ring slots (mod 7 without division)
    if (++sIssue == XRING) sIssue = 0;
    if (++sConv  == XRING) sConv  = 0;
  }

  // ---- final FC: out = h_final @ fc_w^T + fc_b  (h_final in sH[0], wave 0 only) ----
  if (wave == 0) {
    v8f acc = {};
#pragma unroll
    for (int ks = 0; ks < 8; ++ks) {
      v16bf a = lds_frag_a(sH[0], lm, hi, ks * 32);
      v16bf b;
#pragma unroll
      for (int j = 0; j < 16; ++j) {
        const int k = ks * 32 + hi * 16 + j;
        b[j] = (__bf16)fcw[lm * HDIM + k];   // B[k][o] = fc_w[o][k], o = lm
      }
      acc = __builtin_amdgcn_wmma_f32_16x16x32_bf16(false, a, false, b,
                                                    (short)0, acc, false, false);
    }
    const float fb = fcb[lm];
#pragma unroll
    for (int r = 0; r < 8; ++r) {
      out[(size_t)(bb + r + 8 * hi) * ODIM + lm] = acc[r] + fb;
    }
  }
}

extern "C" void kernel_launch(void* const* d_in, const int* in_sizes, int n_in,
                              void* d_out, int out_size, void* d_ws, size_t ws_size,
                              hipStream_t stream) {
  (void)in_sizes; (void)n_in; (void)d_ws; (void)ws_size; (void)out_size;
  const float* x    = (const float*)d_in[0];
  const float* wih  = (const float*)d_in[1];
  const float* whh  = (const float*)d_in[2];
  const float* bh   = (const float*)d_in[3];
  const float* fcw  = (const float*)d_in[4];
  const float* fcb  = (const float*)d_in[5];
  float* out = (float*)d_out;

  // 256 batch rows / 16 per workgroup = 16 blocks; 512 threads = 16 wave32s.
  rnn_fused_kernel<<<16, 512, 0, stream>>>(x, wih, whh, bh, fcw, fcb, out);
}
